// HybridAttention_65481071395837
// MI455X (gfx1250) — matrix operand order repaired and checked by reading, not compiled, not run on hardware
//
#include <hip/hip_runtime.h>

// ---------------------------------------------------------------------------
// Types / helpers
// ---------------------------------------------------------------------------
typedef __bf16 bf16;
typedef __attribute__((ext_vector_type(16))) __bf16 bf16x16;
typedef __attribute__((ext_vector_type(8)))  __bf16 bf16x8;
typedef __attribute__((ext_vector_type(8)))  float  f32x8;

#define EMBED 1024
#define HEADS 16
#define DK    64
#define SEQ   2048
#define BATCH 2
#define MROWS (BATCH * SEQ)   // 4096

static __device__ __forceinline__ bf16 f2bf(float f) { return (bf16)f; }

static __device__ __forceinline__ bf16x16 cat8(bf16x8 lo, bf16x8 hi) {
  return __builtin_shufflevector(lo, hi, 0,1,2,3,4,5,6,7,8,9,10,11,12,13,14,15);
}

static __device__ __forceinline__ f32x8 wmma_bf16(bf16x16 a, bf16x16 b, f32x8 c) {
  return __builtin_amdgcn_wmma_f32_16x16x32_bf16(false, a, false, b, (short)0, c,
                                                 false, false);
}

// ---------------------------------------------------------------------------
// Kernel 1: Q/K/V projections.  y[m,n] = sum_k x[m,k] * W[n,k]  (NT GEMM)
// Block tile 128x128, 8 waves, wave = 64(M) x 32(N), K-step 64.
// Register-staged, ping-pong LDS, one barrier per K-step.
// Q,K stored [B,H,S,Dk]; V stored transposed [B,H,Dk,S].
// ---------------------------------------------------------------------------
__global__ __launch_bounds__(256) void gemm_qkv_kernel(
    const float* __restrict__ x,
    const float* __restrict__ Wq, const float* __restrict__ Wk,
    const float* __restrict__ Wv,
    bf16* __restrict__ Q, bf16* __restrict__ Kb, bf16* __restrict__ Vb)
{
  const float* W;
  bf16* out;
  bool vtrans = false;
  if (blockIdx.z == 0)      { W = Wq; out = Q;  }
  else if (blockIdx.z == 1) { W = Wk; out = Kb; }
  else                      { W = Wv; out = Vb; vtrans = true; }

  __shared__ bf16 At[2][128][64];
  __shared__ bf16 Bt[2][128][64];

  const int tid  = threadIdx.x;
  const int lane = tid & 31;
  const int wave = tid >> 5;
  const int waveM = wave >> 2;   // 0..1
  const int waveN = wave & 3;    // 0..3
  const int hi = lane >> 4, lr = lane & 15;
  const int blockM = blockIdx.y * 128;
  const int blockN = blockIdx.x * 128;

  f32x8 acc[4][2] = {};
  float4 ra[8], rb[8];

  auto load_regs = [&](int k0) {
    #pragma unroll
    for (int i = 0; i < 8; i++) {
      int v4  = tid + i * 256;
      int row = v4 >> 4;
      int kc  = (v4 & 15) * 4;
      ra[i] = *(const float4*)&x[(size_t)(blockM + row) * EMBED + k0 + kc];
      rb[i] = *(const float4*)&W[(size_t)(blockN + row) * EMBED + k0 + kc];
    }
  };

  load_regs(0);
  for (int k0 = 0; k0 < EMBED; k0 += 64) {
    const int buf = (k0 >> 6) & 1;
    #pragma unroll
    for (int i = 0; i < 8; i++) {
      int v4  = tid + i * 256;
      int row = v4 >> 4;
      int kc  = (v4 & 15) * 4;
      At[buf][row][kc + 0] = f2bf(ra[i].x);  At[buf][row][kc + 1] = f2bf(ra[i].y);
      At[buf][row][kc + 2] = f2bf(ra[i].z);  At[buf][row][kc + 3] = f2bf(ra[i].w);
      Bt[buf][row][kc + 0] = f2bf(rb[i].x);  Bt[buf][row][kc + 1] = f2bf(rb[i].y);
      Bt[buf][row][kc + 2] = f2bf(rb[i].z);  Bt[buf][row][kc + 3] = f2bf(rb[i].w);
    }
    __syncthreads();
    if (k0 + 64 < EMBED) load_regs(k0 + 64);   // overlap with WMMAs below
    if (k0 + 128 < EMBED) {                    // prefetch tile after that
      int prow = tid >> 1, pc = (tid & 1) * 32;
      __builtin_prefetch(&x[(size_t)(blockM + prow) * EMBED + k0 + 128 + pc], 0, 0);
      __builtin_prefetch(&W[(size_t)(blockN + prow) * EMBED + k0 + 128 + pc], 0, 0);
    }
    #pragma unroll
    for (int kk = 0; kk < 2; kk++) {
      bf16x16 bfrag[2];
      #pragma unroll
      for (int ns = 0; ns < 2; ns++) {
        int n = waveN * 32 + ns * 16 + lr;
        bfrag[ns] = *(const bf16x16*)&Bt[buf][n][kk * 32 + hi * 16];
      }
      #pragma unroll
      for (int ms = 0; ms < 4; ms++) {
        int m = waveM * 64 + ms * 16 + lr;
        bf16x16 afrag = cat8(*(const bf16x8*)&At[buf][m][kk * 32 + hi * 16],
                             *(const bf16x8*)&At[buf][m][kk * 32 + hi * 16 + 8]);
        #pragma unroll
        for (int ns = 0; ns < 2; ns++)
          acc[ms][ns] = wmma_bf16(afrag, bfrag[ns], acc[ms][ns]);
      }
    }
  }

  // Epilogue: Q,K -> [B,H,S,Dk];  V -> [B,H,Dk,S] (pre-transposed for attn)
  #pragma unroll
  for (int ms = 0; ms < 4; ms++) {
    #pragma unroll
    for (int ns = 0; ns < 2; ns++) {
      #pragma unroll
      for (int r = 0; r < 8; r++) {
        int gm = blockM + waveM * 64 + ms * 16 + hi * 8 + r;   // 0..4095
        int gn = blockN + waveN * 32 + ns * 16 + lr;           // 0..1023
        int b = gm >> 11, s = gm & (SEQ - 1);
        int h = gn >> 6,  d = gn & (DK - 1);
        size_t idx = vtrans
            ? (((size_t)(b * HEADS + h) * DK) + d) * SEQ + s
            : (((size_t)(b * HEADS + h) * SEQ) + s) * DK + d;
        out[idx] = f2bf(acc[ms][ns][r]);
      }
    }
  }
}

// ---------------------------------------------------------------------------
// Kernel 2: per-head flash attention + quantum sin-mixing.
// Workgroup = 8 waves / 128 queries of one (b,h); K/V tiles shared in LDS.
// 64 keys per iteration, ping-pong LDS, one barrier per iteration.
// V arrives pre-transposed as [Dk][S].
// ---------------------------------------------------------------------------
__global__ __launch_bounds__(256) void attn_kernel(
    const bf16* __restrict__ Q, const bf16* __restrict__ K,
    const bf16* __restrict__ V, const float* __restrict__ qw,
    bf16* __restrict__ ctx)
{
  const int bh = blockIdx.y;                 // b*HEADS + h
  const int b  = bh >> 4, h = bh & 15;
  const int tid = threadIdx.x;
  const int lane = tid & 31, wave = tid >> 5;
  const int hi = lane >> 4, lr = lane & 15;
  const int qrow0 = blockIdx.x * 128 + wave * 16;

  const bf16* Qbh  = Q + (size_t)bh * SEQ * DK;
  const bf16* Kbh  = K + (size_t)bh * SEQ * DK;
  const bf16* VbhT = V + (size_t)bh * DK * SEQ;   // [Dk][S]

  __shared__ bf16 Klds[2][64][64];   // K tile (keys x d)
  __shared__ bf16 Vt[2][64][64];     // V tile, transposed (d x keys)
  __shared__ bf16 Pst[8][16][64];    // per-wave P staging

  // Q fragments (registers for whole loop)
  bf16x16 qf[2];
  #pragma unroll
  for (int dh = 0; dh < 2; dh++) {
    const bf16* qp = Qbh + (size_t)(qrow0 + lr) * DK + dh * 32 + hi * 16;
    qf[dh] = cat8(*(const bf16x8*)qp, *(const bf16x8*)(qp + 8));
  }

  const int lrow = tid >> 2;            // 0..63
  const int lcol = (tid & 3) * 16;      // 0,16,32,48
  bf16x8 rk[2], rv[2];
  auto load_tiles = [&](int t0) {
    const bf16* ks = Kbh + (size_t)(t0 + lrow) * DK + lcol;
    rk[0] = *(const bf16x8*)ks;  rk[1] = *(const bf16x8*)(ks + 8);
    const bf16* vs = VbhT + (size_t)lrow * SEQ + t0 + lcol;
    rv[0] = *(const bf16x8*)vs;  rv[1] = *(const bf16x8*)(vs + 8);
  };

  f32x8 o[4] = {};
  float rm[8], rl[8];
  #pragma unroll
  for (int r = 0; r < 8; r++) { rm[r] = -1e30f; rl[r] = 0.0f; }
  const float sc = 0.125f;           // 1/sqrt(DK)

  load_tiles(0);
  for (int t0 = 0; t0 < SEQ; t0 += 64) {
    const int buf = (t0 >> 6) & 1;
    *(bf16x8*)&Klds[buf][lrow][lcol]     = rk[0];
    *(bf16x8*)&Klds[buf][lrow][lcol + 8] = rk[1];
    *(bf16x8*)&Vt[buf][lrow][lcol]       = rv[0];
    *(bf16x8*)&Vt[buf][lrow][lcol + 8]   = rv[1];
    __syncthreads();
    if (t0 + 64 < SEQ) load_tiles(t0 + 64);    // overlap with compute

    // scores for 64 keys: four 16x16 tiles over Dk=64
    f32x8 s[4] = {};
    #pragma unroll
    for (int dh = 0; dh < 2; dh++) {
      #pragma unroll
      for (int j = 0; j < 4; j++) {
        bf16x16 kb = *(const bf16x16*)&Klds[buf][j * 16 + lr][dh * 32 + hi * 16];
        s[j] = wmma_bf16(qf[dh], kb, s[j]);
      }
    }

    // online softmax (rows m = hi*8 + r; stats replicated across 16 lanes)
    float tm[8];
    #pragma unroll
    for (int r = 0; r < 8; r++) {
      s[0][r] *= sc; s[1][r] *= sc; s[2][r] *= sc; s[3][r] *= sc;
      tm[r] = fmaxf(fmaxf(s[0][r], s[1][r]), fmaxf(s[2][r], s[3][r]));
    }
    #pragma unroll
    for (int r = 0; r < 8; r++) {
      #pragma unroll
      for (int off = 1; off < 16; off <<= 1)
        tm[r] = fmaxf(tm[r], __shfl_xor(tm[r], off, 32));
    }
    float alpha[8], ps[8];
    #pragma unroll
    for (int r = 0; r < 8; r++) {
      float mn = fmaxf(rm[r], tm[r]);
      alpha[r] = __expf(rm[r] - mn);
      rm[r] = mn;
      s[0][r] = __expf(s[0][r] - mn);
      s[1][r] = __expf(s[1][r] - mn);
      s[2][r] = __expf(s[2][r] - mn);
      s[3][r] = __expf(s[3][r] - mn);
      ps[r] = (s[0][r] + s[1][r]) + (s[2][r] + s[3][r]);
    }
    #pragma unroll
    for (int r = 0; r < 8; r++) {
      #pragma unroll
      for (int off = 1; off < 16; off <<= 1)
        ps[r] += __shfl_xor(ps[r], off, 32);
      rl[r] = rl[r] * alpha[r] + ps[r];
    }
    #pragma unroll
    for (int dt = 0; dt < 4; dt++)
      #pragma unroll
      for (int r = 0; r < 8; r++) o[dt][r] *= alpha[r];

    // stage P (16x64 bf16) through wave-private LDS to build A fragments
    #pragma unroll
    for (int j = 0; j < 4; j++)
      #pragma unroll
      for (int r = 0; r < 8; r++)
        Pst[wave][hi * 8 + r][j * 16 + lr] = f2bf(s[j][r]);

    #pragma unroll
    for (int tk = 0; tk < 2; tk++) {
      const bf16* pp = &Pst[wave][lr][tk * 32 + hi * 16];
      bf16x16 pa = cat8(*(const bf16x8*)pp, *(const bf16x8*)(pp + 8));
      #pragma unroll
      for (int dt = 0; dt < 4; dt++) {
        bf16x16 vb = *(const bf16x16*)&Vt[buf][dt * 16 + lr][tk * 32 + hi * 16];
        o[dt] = wmma_bf16(pa, vb, o[dt]);
      }
    }
  }

  // epilogue: normalize, quantum mixing, write bf16 ctx [B,S,E]
  const float w = 1.0f / (1.0f + __expf(-qw[h]));
  #pragma unroll
  for (int dt = 0; dt < 4; dt++) {
    #pragma unroll
    for (int r = 0; r < 8; r++) {
      float val = o[dt][r] / rl[r];
      float mixed = w * __sinf(val) + (1.0f - w) * val;
      int s2 = qrow0 + hi * 8 + r;
      int d = dt * 16 + lr;
      ctx[((size_t)(b * SEQ + s2)) * EMBED + h * DK + d] = f2bf(mixed);
    }
  }
}

// ---------------------------------------------------------------------------
// Kernel 3: output projection.  out[m,n] = sum_k ctx[m,k] * Wo[n,k], f32 out.
// Same pipelined structure as kernel 1.
// ---------------------------------------------------------------------------
__global__ __launch_bounds__(256) void gemm_out_kernel(
    const bf16* __restrict__ ctx, const float* __restrict__ Wo,
    float* __restrict__ outp)
{
  __shared__ bf16 At[2][128][64];
  __shared__ bf16 Bt[2][128][64];

  const int tid  = threadIdx.x;
  const int lane = tid & 31;
  const int wave = tid >> 5;
  const int waveM = wave >> 2;
  const int waveN = wave & 3;
  const int hi = lane >> 4, lr = lane & 15;
  const int blockM = blockIdx.y * 128;
  const int blockN = blockIdx.x * 128;

  f32x8 acc[4][2] = {};
  bf16x8 rcx[4];
  float4 rb[8];

  auto load_regs = [&](int k0) {
    #pragma unroll
    for (int i = 0; i < 4; i++) {          // A: bf16 copy
      int v8  = tid + i * 256;
      int row = v8 >> 3;
      int kc  = (v8 & 7) * 8;
      rcx[i] = *(const bf16x8*)&ctx[(size_t)(blockM + row) * EMBED + k0 + kc];
    }
    #pragma unroll
    for (int i = 0; i < 8; i++) {          // B: f32
      int v4  = tid + i * 256;
      int row = v4 >> 4;
      int kc  = (v4 & 15) * 4;
      rb[i] = *(const float4*)&Wo[(size_t)(blockN + row) * EMBED + k0 + kc];
    }
  };

  load_regs(0);
  for (int k0 = 0; k0 < EMBED; k0 += 64) {
    const int buf = (k0 >> 6) & 1;
    #pragma unroll
    for (int i = 0; i < 4; i++) {
      int v8  = tid + i * 256;
      int row = v8 >> 3;
      int kc  = (v8 & 7) * 8;
      *(bf16x8*)&At[buf][row][kc] = rcx[i];
    }
    #pragma unroll
    for (int i = 0; i < 8; i++) {
      int v4  = tid + i * 256;
      int row = v4 >> 4;
      int kc  = (v4 & 15) * 4;
      Bt[buf][row][kc + 0] = f2bf(rb[i].x);  Bt[buf][row][kc + 1] = f2bf(rb[i].y);
      Bt[buf][row][kc + 2] = f2bf(rb[i].z);  Bt[buf][row][kc + 3] = f2bf(rb[i].w);
    }
    __syncthreads();
    if (k0 + 64 < EMBED) load_regs(k0 + 64);
    if (k0 + 128 < EMBED) {
      int prow = tid >> 1, pc = (tid & 1) * 32;
      __builtin_prefetch(&ctx[(size_t)(blockM + prow) * EMBED + k0 + 128 + pc], 0, 0);
      __builtin_prefetch(&Wo[(size_t)(blockN + prow) * EMBED + k0 + 128 + pc], 0, 0);
    }
    #pragma unroll
    for (int kk = 0; kk < 2; kk++) {
      bf16x16 bfrag[2];
      #pragma unroll
      for (int ns = 0; ns < 2; ns++) {
        int n = waveN * 32 + ns * 16 + lr;
        bfrag[ns] = *(const bf16x16*)&Bt[buf][n][kk * 32 + hi * 16];
      }
      #pragma unroll
      for (int ms = 0; ms < 4; ms++) {
        int m = waveM * 64 + ms * 16 + lr;
        bf16x16 afrag = cat8(*(const bf16x8*)&At[buf][m][kk * 32 + hi * 16],
                             *(const bf16x8*)&At[buf][m][kk * 32 + hi * 16 + 8]);
        #pragma unroll
        for (int ns = 0; ns < 2; ns++)
          acc[ms][ns] = wmma_bf16(afrag, bfrag[ns], acc[ms][ns]);
      }
    }
  }

  #pragma unroll
  for (int ms = 0; ms < 4; ms++)
    #pragma unroll
    for (int ns = 0; ns < 2; ns++)
      #pragma unroll
      for (int r = 0; r < 8; r++) {
        int gm = blockM + waveM * 64 + ms * 16 + hi * 8 + r;
        int gn = blockN + waveN * 32 + ns * 16 + lr;
        outp[(size_t)gm * EMBED + gn] = acc[ms][ns][r];
      }
}

// ---------------------------------------------------------------------------
// Launch
// ---------------------------------------------------------------------------
extern "C" void kernel_launch(void* const* d_in, const int* in_sizes, int n_in,
                              void* d_out, int out_size, void* d_ws, size_t ws_size,
                              hipStream_t stream) {
  const float* x  = (const float*)d_in[0];
  const float* Wq = (const float*)d_in[1];
  const float* Wk = (const float*)d_in[2];
  const float* Wv = (const float*)d_in[3];
  const float* Wo = (const float*)d_in[4];
  const float* qw = (const float*)d_in[5];

  const size_t NBH = (size_t)MROWS * EMBED;   // 4M elements
  bf16* Qb  = (bf16*)d_ws;
  bf16* Kb  = Qb + NBH;
  bf16* Vb  = Kb + NBH;
  bf16* ctx = Vb + NBH;

  dim3 gridP(EMBED / 128, MROWS / 128, 3);
  gemm_qkv_kernel<<<gridP, 256, 0, stream>>>(x, Wq, Wk, Wv, Qb, Kb, Vb);

  dim3 gridA(SEQ / 128, BATCH * HEADS);
  attn_kernel<<<gridA, 256, 0, stream>>>(Qb, Kb, Vb, qw, ctx);

  dim3 gridO(EMBED / 128, MROWS / 128);
  gemm_out_kernel<<<gridO, 256, 0, stream>>>(ctx, Wo, (float*)d_out);
}
